// VariationalGraohAutoEncoder_61194694033656
// MI455X (gfx1250) — matrix operand order repaired and checked by reading, not compiled, not run on hardware
//
#include <hip/hip_runtime.h>
#include <hip/hip_bf16.h>

// VGAE forward for MI455X (gfx1250).
// f16 WMMA GEMMs (f32 accumulate) with fragment-prepacked f16 weights and f16
// activation buffers; f32 atomic scatters for the edge segment-sums.

typedef __attribute__((ext_vector_type(16))) _Float16 v16h;
typedef __attribute__((ext_vector_type(8)))  _Float16 v8h;
typedef __attribute__((ext_vector_type(8)))  float    v8f;

__device__ __forceinline__ v8f wmma32(const v16h& a, const v16h& b, const v8f& c) {
  // D = A(16x32 f16) x B(32x16 f16) + C(16x16 f32)
  return __builtin_amdgcn_wmma_f32_16x16x32_f16(false, a, false, b, (short)0, c,
                                                false, false);
}

// A fragment (16x32 f16) from a row-major f16 matrix. rowk = &A[m][k0] for this
// lane's row m. Per lane: elements 0..7 = K[khalf..khalf+7], 8..15 = K[16+khalf..].
// Two contiguous 16B chunks -> 2x global_load_b128.
__device__ __forceinline__ v16h load_a_f16(const _Float16* __restrict__ rowk) {
  const int khalf = (((int)threadIdx.x & 31) >> 4) << 3;  // 0 or 8
  v8h lo = *(const v8h*)(rowk + khalf);
  v8h hi = *(const v8h*)(rowk + 16 + khalf);
  return __builtin_shufflevector(lo, hi, 0, 1, 2, 3, 4, 5, 6, 7,
                                 8, 9, 10, 11, 12, 13, 14, 15);
}

// B fragment from fragment-order prepacked weights: one contiguous 32B read/lane.
__device__ __forceinline__ v16h load_b_packed(const _Float16* __restrict__ Bp,
                                              int nct, int kt, int ct) {
  const int lane = (int)threadIdx.x & 31;
  return *(const v16h*)(Bp + ((((size_t)kt * nct + ct) * 32 + lane) << 4));
}

// ---------------- prep kernels ----------------

__global__ void zero_f32(float* __restrict__ p, size_t n) {
  size_t stride = (size_t)gridDim.x * blockDim.x;
  for (size_t i = (size_t)blockIdx.x * blockDim.x + threadIdx.x; i < n; i += stride)
    p[i] = 0.0f;
}

__global__ void cvt_f16_kernel(const float* __restrict__ in, _Float16* __restrict__ out,
                               size_t n) {
  size_t stride = (size_t)gridDim.x * blockDim.x;
  for (size_t i = (size_t)blockIdx.x * blockDim.x + threadIdx.x; i < n; i += stride)
    out[i] = (_Float16)in[i];
}

// Pack row-major f32 B[K x NC] into WMMA-B fragment order, f16:
// out[(((kt*nct + ct)*32 + lane)*16 + e)] = B[kt*32 + (e>>3)*16 + (lane>>4)*8 + (e&7)][ct*16 + (lane&15)]
__global__ void pack_b_kernel(const float* __restrict__ B, _Float16* __restrict__ out,
                              int K, int NC) {
  int total = K * NC;
  int idx = blockIdx.x * blockDim.x + threadIdx.x;
  if (idx >= total) return;
  const int nct = NC >> 4;
  const int e = idx & 15;
  const int lane = (idx >> 4) & 31;
  const int rest = idx >> 9;           // kt*nct + ct
  const int ct = rest % nct;
  const int kt = rest / nct;
  const int k = kt * 32 + ((e >> 3) << 4) + ((lane >> 4) << 3) + (e & 7);
  const int n = ct * 16 + (lane & 15);
  out[idx] = (_Float16)B[(size_t)k * NC + n];
}

__global__ void count_deg_kernel(const int* __restrict__ dst, float* __restrict__ deg, int E) {
  int e = blockIdx.x * blockDim.x + threadIdx.x;
  if (e < E) atomicAdd(&deg[dst[e]], 1.0f);
}

// out[dst[e], :] += feat[src[e], :] — one wave per edge; f16 gather, f32 accumulate.
__global__ void scatter_add_rows_f16(const _Float16* __restrict__ feat, float* __restrict__ out,
                                     const int* __restrict__ src, const int* __restrict__ dst,
                                     int E, int d) {
  int e = blockIdx.x * (blockDim.x >> 5) + ((int)threadIdx.x >> 5);
  if (e >= E) return;
  const _Float16* fs = feat + (size_t)src[e] * d;
  float* fo = out + (size_t)dst[e] * d;
  for (int f = (int)threadIdx.x & 31; f < d; f += 32)
    atomicAdd(&fo[f], (float)fs[f]);
}

// aggh[i,:] = f16( acc[i,:] / max(deg[i],1) )
__global__ void norm_rows_f16(const float* __restrict__ acc, const float* __restrict__ deg,
                              _Float16* __restrict__ out, int N, int dshift) {
  size_t n = (size_t)N << dshift;
  size_t stride = (size_t)gridDim.x * blockDim.x;
  for (size_t i = (size_t)blockIdx.x * blockDim.x + threadIdx.x; i < n; i += stride) {
    int row = (int)(i >> dshift);
    out[i] = (_Float16)(acc[i] * (1.0f / fmaxf(deg[row], 1.0f)));
  }
}

// ---------------- WMMA GEMM kernels (one wave = one 16x16 C tile) ----------------

// h1 = relu(agg@W1_l + x@W1_r + b1) + pad(x)  -> h1h (f16).  K=128, out width 256.
__global__ void sage1_kernel(const _Float16* __restrict__ aggh, const _Float16* __restrict__ xh,
                             const float* __restrict__ x,
                             const _Float16* __restrict__ W1lp, const _Float16* __restrict__ W1rp,
                             const float* __restrict__ b, _Float16* __restrict__ h1h, int N) {
  const int row0 = blockIdx.x << 4;
  const int ct = blockIdx.y;             // 16 col tiles
  const int lane = (int)threadIdx.x & 31;
  int mA = row0 + (lane & 15);
  if (mA >= N) mA = N - 1;
  v8f acc = {};
#pragma unroll
  for (int kt = 0; kt < 4; ++kt) {
    acc = wmma32(load_a_f16(aggh + (size_t)mA * 128 + kt * 32), load_b_packed(W1lp, 16, kt, ct), acc);
    acc = wmma32(load_a_f16(xh   + (size_t)mA * 128 + kt * 32), load_b_packed(W1rp, 16, kt, ct), acc);
  }
  const int n = (ct << 4) + (lane & 15);
  const float bias = b[n];
  const int mbase = row0 + ((lane >> 4) << 3);
#pragma unroll
  for (int r = 0; r < 8; ++r) {
    int m = mbase + r;
    if (m < N) {
      float v = fmaxf(acc[r] + bias, 0.0f);
      if (n < 128) v += x[(size_t)m * 128 + n];  // fit_and_add zero-pad residual
      h1h[(size_t)m * 256 + n] = (_Float16)v;
    }
  }
}

// h2 = relu(agg2@W2_l + h1@W2_r + b2) + (x@Wr + br)  -> h2h (f16).
__global__ void sage2_kernel(const _Float16* __restrict__ aggh, const _Float16* __restrict__ h1h,
                             const _Float16* __restrict__ xh,
                             const _Float16* __restrict__ W2lp, const _Float16* __restrict__ W2rp,
                             const float* __restrict__ b2,
                             const _Float16* __restrict__ Wrp, const float* __restrict__ br,
                             _Float16* __restrict__ h2h, int N) {
  const int row0 = blockIdx.x << 4;
  const int ct = blockIdx.y;
  const int lane = (int)threadIdx.x & 31;
  int mA = row0 + (lane & 15);
  if (mA >= N) mA = N - 1;
  v8f acc = {};
#pragma unroll
  for (int kt = 0; kt < 8; ++kt) {
    acc = wmma32(load_a_f16(aggh + (size_t)mA * 256 + kt * 32), load_b_packed(W2lp, 16, kt, ct), acc);
    acc = wmma32(load_a_f16(h1h  + (size_t)mA * 256 + kt * 32), load_b_packed(W2rp, 16, kt, ct), acc);
  }
  v8f acc2 = {};
#pragma unroll
  for (int kt = 0; kt < 4; ++kt)
    acc2 = wmma32(load_a_f16(xh + (size_t)mA * 128 + kt * 32), load_b_packed(Wrp, 16, kt, ct), acc2);
  const int n = (ct << 4) + (lane & 15);
  const float bb2 = b2[n], bbr = br[n];
  const int mbase = row0 + ((lane >> 4) << 3);
#pragma unroll
  for (int r = 0; r < 8; ++r) {
    int m = mbase + r;
    if (m < N)
      h2h[(size_t)m * 256 + n] = (_Float16)(fmaxf(acc[r] + bb2, 0.0f) + acc2[r] + bbr);
  }
}

// xw[:,0:64] = h2@Wmu ; xw[:,64:128] = h2@Wls  (f32 out for GCN epilogue). K=256.
__global__ void gcn_xw_kernel(const _Float16* __restrict__ h2h,
                              const _Float16* __restrict__ Wmup, const _Float16* __restrict__ Wlsp,
                              float* __restrict__ xw, int N) {
  const int row0 = blockIdx.x << 4;
  const int colT = blockIdx.y;  // 0..7
  const _Float16* Wp = (colT < 4) ? Wmup : Wlsp;
  const int ct = colT & 3;      // nct = 4 within each 64-wide head
  const int lane = (int)threadIdx.x & 31;
  int mA = row0 + (lane & 15);
  if (mA >= N) mA = N - 1;
  v8f acc = {};
#pragma unroll
  for (int kt = 0; kt < 8; ++kt)
    acc = wmma32(load_a_f16(h2h + (size_t)mA * 256 + kt * 32), load_b_packed(Wp, 4, kt, ct), acc);
  const int n = (colT << 4) + (lane & 15);
  const int mbase = row0 + ((lane >> 4) << 3);
#pragma unroll
  for (int r = 0; r < 8; ++r) {
    int m = mbase + r;
    if (m < N) xw[(size_t)m * 128 + n] = acc[r];
  }
}

// GCN self-loop term + bias: fully initializes d_out (mu || logstd).
__global__ void gcn_init_kernel(const float* __restrict__ xw, const float* __restrict__ deg,
                                const float* __restrict__ bmu, const float* __restrict__ bls,
                                float* __restrict__ out, int N) {
  size_t n = (size_t)N * 64;
  size_t stride = (size_t)gridDim.x * blockDim.x;
  for (size_t i = (size_t)blockIdx.x * blockDim.x + threadIdx.x; i < n; i += stride) {
    int row = (int)(i >> 6);
    int f = (int)(i & 63);
    float inv = 1.0f / (deg[row] + 1.0f);  // dinv*dinv of self loop
    out[i]     = xw[(size_t)row * 128 + f]      * inv + bmu[f];  // mu
    out[n + i] = xw[(size_t)row * 128 + 64 + f] * inv + bls[f];  // logstd
  }
}

// Edge term: out[dst] += xw[src] * rsqrt(deg[src]+1) * rsqrt(deg[dst]+1)
__global__ void gcn_scatter_kernel(const float* __restrict__ xw, const float* __restrict__ deg,
                                   const int* __restrict__ src, const int* __restrict__ dst,
                                   float* __restrict__ out, int N, int E) {
  int e = blockIdx.x * (blockDim.x >> 5) + ((int)threadIdx.x >> 5);
  if (e >= E) return;
  const int s = src[e], t = dst[e];
  const float coef = rsqrtf(deg[s] + 1.0f) * rsqrtf(deg[t] + 1.0f);
  const float* xs = xw + (size_t)s * 128;
  float* omu = out + (size_t)t * 64;
  float* ols = out + (size_t)N * 64 + (size_t)t * 64;
  const int lane = (int)threadIdx.x & 31;
#pragma unroll
  for (int f0 = 0; f0 < 64; f0 += 32) {
    int f = f0 + lane;
    atomicAdd(&omu[f], xs[f] * coef);
    atomicAdd(&ols[f], xs[64 + f] * coef);
  }
}

// ---------------- launch ----------------

extern "C" void kernel_launch(void* const* d_in, const int* in_sizes, int n_in,
                              void* d_out, int out_size, void* d_ws, size_t ws_size,
                              hipStream_t stream) {
  (void)n_in; (void)out_size; (void)ws_size;
  const float* x   = (const float*)d_in[0];
  const int*   ei  = (const int*)d_in[1];
  const float* W1l = (const float*)d_in[2];
  const float* W1r = (const float*)d_in[3];
  const float* b1  = (const float*)d_in[4];
  const float* W2l = (const float*)d_in[5];
  const float* W2r = (const float*)d_in[6];
  const float* b2  = (const float*)d_in[7];
  const float* Wr  = (const float*)d_in[8];
  const float* br  = (const float*)d_in[9];
  const float* Wmu = (const float*)d_in[10];
  const float* bmu = (const float*)d_in[11];
  const float* Wls = (const float*)d_in[12];
  const float* bls = (const float*)d_in[13];

  const int N = in_sizes[0] / 128;
  const int E = in_sizes[1] / 2;
  const int* src = ei;
  const int* dst = ei + E;

  // ---- workspace layout (256B aligned regions) ----
  auto al = [](size_t v) { return (v + 255) & ~(size_t)255; };
  char* p = (char*)d_ws;
  size_t off = 0;
  float* deg = (float*)(p + off);        off = al(off + (size_t)N * 4);
  float* accum = (float*)(p + off);      off = al(off + (size_t)N * 256 * 4);  // agg1/agg2 accum; later xw
  _Float16* xh   = (_Float16*)(p + off); off = al(off + (size_t)N * 128 * 2);
  _Float16* aggh = (_Float16*)(p + off); off = al(off + (size_t)N * 256 * 2);
  _Float16* h1h  = (_Float16*)(p + off); off = al(off + (size_t)N * 256 * 2);
  _Float16* h2h  = (_Float16*)(p + off); off = al(off + (size_t)N * 256 * 2);
  _Float16* W1lp = (_Float16*)(p + off); off += (size_t)128 * 256 * 2;
  _Float16* W1rp = (_Float16*)(p + off); off += (size_t)128 * 256 * 2;
  _Float16* W2lp = (_Float16*)(p + off); off += (size_t)256 * 256 * 2;
  _Float16* W2rp = (_Float16*)(p + off); off += (size_t)256 * 256 * 2;
  _Float16* Wrp  = (_Float16*)(p + off); off += (size_t)128 * 256 * 2;
  _Float16* Wmup = (_Float16*)(p + off); off += (size_t)256 * 64 * 2;
  _Float16* Wlsp = (_Float16*)(p + off); off += (size_t)256 * 64 * 2;
  float* xw = accum;                     // reuse after agg2 is consumed
  float* out = (float*)d_out;

  const dim3 gsGrid(4096);
  const int rowT = (N + 15) / 16;
  const int edgeBlocks = (E + 7) / 8;    // wave per edge, 8 waves per 256-thread block

  // degrees + input conversion + weight packing
  zero_f32<<<gsGrid, 256, 0, stream>>>(deg, (size_t)N);
  count_deg_kernel<<<(E + 255) / 256, 256, 0, stream>>>(dst, deg, E);
  cvt_f16_kernel<<<gsGrid, 256, 0, stream>>>(x, xh, (size_t)N * 128);
  pack_b_kernel<<<(128 * 256 + 255) / 256, 256, 0, stream>>>(W1l, W1lp, 128, 256);
  pack_b_kernel<<<(128 * 256 + 255) / 256, 256, 0, stream>>>(W1r, W1rp, 128, 256);
  pack_b_kernel<<<(256 * 256 + 255) / 256, 256, 0, stream>>>(W2l, W2lp, 256, 256);
  pack_b_kernel<<<(256 * 256 + 255) / 256, 256, 0, stream>>>(W2r, W2rp, 256, 256);
  pack_b_kernel<<<(128 * 256 + 255) / 256, 256, 0, stream>>>(Wr, Wrp, 128, 256);
  pack_b_kernel<<<(256 * 64 + 255) / 256, 256, 0, stream>>>(Wmu, Wmup, 256, 64);
  pack_b_kernel<<<(256 * 64 + 255) / 256, 256, 0, stream>>>(Wls, Wlsp, 256, 64);

  // layer 1: mean aggregate x, SAGE GEMM (+relu, +pad residual)
  zero_f32<<<gsGrid, 256, 0, stream>>>(accum, (size_t)N * 128);
  scatter_add_rows_f16<<<edgeBlocks, 256, 0, stream>>>(xh, accum, src, dst, E, 128);
  norm_rows_f16<<<gsGrid, 256, 0, stream>>>(accum, deg, aggh, N, 7);
  sage1_kernel<<<dim3(rowT, 16), 32, 0, stream>>>(aggh, xh, x, W1lp, W1rp, b1, h1h, N);

  // layer 2: mean aggregate h1, SAGE GEMM (+relu) + x@Wr residual
  zero_f32<<<gsGrid, 256, 0, stream>>>(accum, (size_t)N * 256);
  scatter_add_rows_f16<<<edgeBlocks, 256, 0, stream>>>(h1h, accum, src, dst, E, 256);
  norm_rows_f16<<<gsGrid, 256, 0, stream>>>(accum, deg, aggh, N, 8);
  sage2_kernel<<<dim3(rowT, 16), 32, 0, stream>>>(aggh, h1h, xh, W2lp, W2rp, b2, Wrp, br, h2h, N);

  // GCN heads: xw = h2 @ [Wmu | Wls], self term + normalized edge scatter
  gcn_xw_kernel<<<dim3(rowT, 8), 32, 0, stream>>>(h2h, Wmup, Wlsp, xw, N);
  gcn_init_kernel<<<gsGrid, 256, 0, stream>>>(xw, deg, bmu, bls, out, N);
  gcn_scatter_kernel<<<edgeBlocks, 256, 0, stream>>>(xw, deg, src, dst, out, N, E);
}
